// GCUTPL_50173807952233
// MI455X (gfx1250) — matrix operand order repaired and checked by reading, compiled, not run on hardware
//
#include <hip/hip_runtime.h>
#include <hip/hip_bf16.h>

// ---------------------------------------------------------------------------
// Fused GNN edge-MLP for MI455X (gfx1250, wave32, WMMA).
//   rows = E + N self loops; h = [x_i | x_j - x_i]  (256 features)
//   h = BN(relu(h W1^T + b1)); h = BN(relu(h W2^T + b2))
//   agg = segment_max(h, tgt);  out = BN(relu(agg W3^T + b3))
// Compute ~175 GFLOP, traffic ~100 MB (all L2-resident) -> matrix-unit bound.
// bf16 WMMA (16x16x32) with f32 accumulation; activations live in LDS/regs.
// ---------------------------------------------------------------------------

#define C_IN   128
#define CK     256           // K dim for every layer (= 2*C_IN = C_OUT)
#define TILE_M 128           // rows per workgroup
#define BLOCK  256           // 8 waves
#define LROW   512           // LDS bytes per row (256 bf16)
#define BN_EPS 1e-5f

typedef __bf16 bf16;
typedef __attribute__((ext_vector_type(16))) __bf16 v16bf;
typedef __attribute__((ext_vector_type(8)))  float  v8f;

static __device__ __forceinline__ unsigned short bfbits(float f) {
  bf16 h = (bf16)f;
  return __builtin_bit_cast(unsigned short, h);
}
// Order-preserving float -> uint key so GLOBAL_ATOMIC_MAX_U32 == float max.
static __device__ __forceinline__ unsigned encKey(float f) {
  unsigned u = __builtin_bit_cast(unsigned, f);
  return ((int)u >= 0) ? (u | 0x80000000u) : ~u;
}
static __device__ __forceinline__ float decKey(unsigned k) {
  unsigned u = (k & 0x80000000u) ? (k ^ 0x80000000u) : ~k;
  return __builtin_bit_cast(float, u);
}
#define ENC_NEG_INF 0x007FFFFFu   // encKey(-inf)

// --- swizzled LDS tile helpers -------------------------------------------
// Row stride is 512B; 16-byte chunks are XOR-swizzled by the row so the 16
// lanes of an A-fragment ds_load_b128 hit 16 different bank groups.
static __device__ __forceinline__ void stBf4(unsigned short* sm, int row, int col,
                                             float a, float b, float c, float d) {
  int ch   = col >> 3;
  int phys = (ch & 16) | ((ch ^ row) & 15);
  unsigned lo = ((unsigned)bfbits(b) << 16) | bfbits(a);
  unsigned hi = ((unsigned)bfbits(d) << 16) | bfbits(c);
  *(unsigned long long*)((char*)sm + row * LROW + phys * 16 + (col & 7) * 2) =
      ((unsigned long long)hi << 32) | lo;
}
static __device__ __forceinline__ void stBf1(unsigned short* sm, int row, int col, float f) {
  int ch   = col >> 3;
  int phys = (ch & 16) | ((ch ^ row) & 15);
  *(unsigned short*)((char*)sm + row * LROW + phys * 16 + (col & 7) * 2) = bfbits(f);
}

// A-fragment (16x32 bf16): lanes 0-15 -> K 0..15, lanes 16-31 -> K 16..31.
static __device__ __forceinline__ v16bf ldsA(const unsigned short* sm, int row, int kc, int lane) {
  int h   = (lane & 16) ? 1 : 0;
  int ch0 = kc * 4 + 2 * h;
  int ch1 = ch0 + 1;
  int p0  = (ch0 & 16) | ((ch0 ^ row) & 15);
  int p1  = (ch1 & 16) | ((ch1 ^ row) & 15);
  const char* base = (const char*)sm + row * LROW;
  union { uint4 q[2]; v16bf v; } u;
  u.q[0] = *(const uint4*)(base + p0 * 16);
  u.q[1] = *(const uint4*)(base + p1 * 16);
  return u.v;
}
// B-fragment: prepacked so each lane's 32 bytes are contiguous.
static __device__ __forceinline__ v16bf ldB(const unsigned short* Wp, int kc, int n, int lane) {
  const char* p = (const char*)Wp + (((kc << 4) + n) << 10) + (lane << 5);
  union { uint4 q[2]; v16bf v; } u;
  u.q[0] = *(const uint4*)p;
  u.q[1] = *(const uint4*)(p + 16);
  return u.v;
}

// One wave: C[16 x 256] = A[16 x 256] * B[256 x 256]   (bf16 -> f32 WMMA)
static __device__ __forceinline__ void gemm16x256(const unsigned short* sm, int rowBase,
                                                  const unsigned short* Wp, int lane,
                                                  v8f acc[16]) {
  const v8f zero = {0.f, 0.f, 0.f, 0.f, 0.f, 0.f, 0.f, 0.f};
#pragma unroll
  for (int n = 0; n < 16; ++n) acc[n] = zero;
  int row = rowBase + (lane & 15);
  for (int kc = 0; kc < 8; ++kc) {
    v16bf a = ldsA(sm, row, kc, lane);
#pragma unroll
    for (int n = 0; n < 16; ++n) {
      v16bf b = ldB(Wp, kc, n, lane);
      acc[n] = __builtin_amdgcn_wmma_f32_16x16x32_bf16(
          /*neg_a=*/false, a, /*neg_b=*/false, b,
          /*c_mod=*/(short)0, acc[n], /*reuse_a=*/false, /*reuse_b=*/false);
    }
  }
}

// --- weight repack: fp32 [O][K] -> bf16 B-fragment order ------------------
__global__ void pack_weights(const float* __restrict__ W1, const float* __restrict__ W2,
                             const float* __restrict__ W3, unsigned short* __restrict__ out) {
  int t = blockIdx.x * blockDim.x + threadIdx.x;   // (weight, tile, lane)
  if (t >= 3 * 128 * 32) return;
  int w    = t / (128 * 32);
  int rem  = t % (128 * 32);
  int tile = rem >> 5;            // kc*16 + n
  int lane = rem & 31;
  int kc = tile >> 4, n = tile & 15;
  const float* W = (w == 0) ? W1 : (w == 1) ? W2 : W3;
  int col   = n * 16 + (lane & 15);
  int kbase = kc * 32 + ((lane & 16) ? 16 : 0);
  unsigned* dst = (unsigned*)(out + (size_t)w * 65536 + tile * 512 + lane * 16);
#pragma unroll
  for (int v = 0; v < 8; ++v) {
    float a = W[col * CK + kbase + 2 * v];
    float b = W[col * CK + kbase + 2 * v + 1];
    dst[v] = ((unsigned)bfbits(b) << 16) | bfbits(a);
  }
}

__global__ void init_agg(uint4* __restrict__ agg, long long n4) {
  long long i = blockIdx.x * (long long)blockDim.x + threadIdx.x;
  if (i < n4) agg[i] = make_uint4(ENC_NEG_INF, ENC_NEG_INF, ENC_NEG_INF, ENC_NEG_INF);
}

// --- fused edge MLP (layers 1+2) + scatter-max ----------------------------
__global__ __launch_bounds__(BLOCK) void edge_mlp(
    const float* __restrict__ x, const long long* __restrict__ ei, long long E,
    long long totRows,
    const unsigned short* __restrict__ W1p, const unsigned short* __restrict__ W2p,
    const float* __restrict__ b1, const float* __restrict__ g1, const float* __restrict__ be1,
    const float* __restrict__ m1, const float* __restrict__ v1,
    const float* __restrict__ b2, const float* __restrict__ g2, const float* __restrict__ be2,
    const float* __restrict__ m2, const float* __restrict__ v2,
    unsigned* __restrict__ agg) {
  __shared__ __align__(16) unsigned short sm[TILE_M * CK];   // 64 KB bf16 tile
  const int tid = threadIdx.x, lane = tid & 31, wid = tid >> 5;
  const long long rowbase = (long long)blockIdx.x * TILE_M;

  // Stage A1 = [x_i | x_j - x_i] as bf16 (2 threads per row, 64 cols each).
  {
    int r = tid >> 1, half = tid & 1;
    long long row = rowbase + r;
    long long si = 0, ti = 0;
    if (row < totRows) {
      if (row < E) { si = ei[row]; ti = ei[E + row]; }
      else         { si = ti = row - E; }
    }
    const float* xi = x + ti * C_IN;
    const float* xj = x + si * C_IN;
    int c0 = half * 64;
    for (int c = c0; c < c0 + 64; c += 4) {
      float4 a = *(const float4*)(xi + c);
      float4 b = *(const float4*)(xj + c);
      stBf4(sm, r, c, a.x, a.y, a.z, a.w);
      stBf4(sm, r, C_IN + c, b.x - a.x, b.y - a.y, b.z - a.z, b.w - a.w);
    }
  }
  // Per-lane aggregation target for this wave's 16 rows (lanes 0-15 valid).
  int myTgt = 0;
  {
    long long rr = rowbase + wid * 16 + (lane & 15);
    if (rr < totRows) myTgt = (int)((rr < E) ? ei[E + rr] : (rr - E));
  }
  __syncthreads();

  const int rowB = wid * 16;
  v8f acc[16];

  // Layer 1
  gemm16x256(sm, rowB, W1p, lane, acc);
#pragma unroll
  for (int n = 0; n < 16; ++n) {
    int col = n * 16 + (lane & 15);
    float bias = b1[col];
    float s = g1[col] * rsqrtf(v1[col] + BN_EPS);
    float t = be1[col] - m1[col] * s;
#pragma unroll
    for (int p = 0; p < 8; ++p) {
      float z = fmaxf(acc[n][p] + bias, 0.f) * s + t;
      stBf1(sm, rowB + p + ((lane & 16) ? 8 : 0), col, z);   // wave-private rows
    }
  }

  // Layer 2 (reads the rows this wave just wrote; DS is in-order per wave)
  gemm16x256(sm, rowB, W2p, lane, acc);
#pragma unroll
  for (int n = 0; n < 16; ++n) {
    int col = n * 16 + (lane & 15);
    float bias = b2[col];
    float s = g2[col] * rsqrtf(v2[col] + BN_EPS);
    float t = be2[col] - m2[col] * s;
#pragma unroll
    for (int p = 0; p < 8; ++p) {
      float z = fmaxf(acc[n][p] + bias, 0.f) * s + t;
      int m = p + ((lane & 16) ? 8 : 0);
      int tg = __shfl(myTgt, m, 32);
      long long grow = rowbase + rowB + m;
      if (grow < totRows)
        atomicMax(&agg[(long long)tg * CK + col], encKey(z));  // float max via key order
    }
  }
}

// --- final layer on aggregated node features ------------------------------
__global__ __launch_bounds__(BLOCK) void final_mlp(
    const unsigned* __restrict__ agg, const unsigned short* __restrict__ W3p,
    const float* __restrict__ b3, const float* __restrict__ g3, const float* __restrict__ be3,
    const float* __restrict__ m3, const float* __restrict__ v3,
    float* __restrict__ out, int N) {
  __shared__ __align__(16) unsigned short sm[TILE_M * CK];
  const int tid = threadIdx.x, lane = tid & 31, wid = tid >> 5;
  const int rowbase = blockIdx.x * TILE_M;

  {
    int r = tid >> 1, half = tid & 1;
    int row = rowbase + r;
    int rsafe = (row < N) ? row : 0;
    const unsigned* src = agg + (size_t)rsafe * CK;
    int c0 = half * 128;
    for (int c = c0; c < c0 + 128; c += 4) {
      uint4 k = *(const uint4*)(src + c);
      stBf4(sm, r, c, decKey(k.x), decKey(k.y), decKey(k.z), decKey(k.w));
    }
  }
  __syncthreads();

  v8f acc[16];
  gemm16x256(sm, wid * 16, W3p, lane, acc);
#pragma unroll
  for (int n = 0; n < 16; ++n) {
    int col = n * 16 + (lane & 15);
    float bias = b3[col];
    float s = g3[col] * rsqrtf(v3[col] + BN_EPS);
    float t = be3[col] - m3[col] * s;
#pragma unroll
    for (int p = 0; p < 8; ++p) {
      float z = fmaxf(acc[n][p] + bias, 0.f) * s + t;
      int grow = rowbase + wid * 16 + p + ((lane & 16) ? 8 : 0);
      if (grow < N) out[(size_t)grow * CK + col] = z;
    }
  }
}

extern "C" void kernel_launch(void* const* d_in, const int* in_sizes, int n_in,
                              void* d_out, int out_size, void* d_ws, size_t ws_size,
                              hipStream_t stream) {
  const float*     x  = (const float*)d_in[0];
  const long long* ei = (const long long*)d_in[1];
  const float *W1 = (const float*)d_in[2],  *b1 = (const float*)d_in[3],
              *g1 = (const float*)d_in[4],  *be1 = (const float*)d_in[5],
              *m1 = (const float*)d_in[6],  *v1 = (const float*)d_in[7];
  const float *W2 = (const float*)d_in[8],  *b2 = (const float*)d_in[9],
              *g2 = (const float*)d_in[10], *be2 = (const float*)d_in[11],
              *m2 = (const float*)d_in[12], *v2 = (const float*)d_in[13];
  const float *W3 = (const float*)d_in[14], *b3 = (const float*)d_in[15],
              *g3 = (const float*)d_in[16], *be3 = (const float*)d_in[17],
              *m3 = (const float*)d_in[18], *v3 = (const float*)d_in[19];

  const int       N       = in_sizes[0] / C_IN;
  const long long E       = in_sizes[1] / 2;
  const long long totRows = E + (long long)N;

  unsigned short* Wp  = (unsigned short*)d_ws;                 // 3 x 128 KB bf16
  unsigned*       agg = (unsigned*)((char*)d_ws + 3 * 131072); // N x 256 keys

  pack_weights<<<(3 * 128 * 32 + 255) / 256, 256, 0, stream>>>(W1, W2, W3, Wp);

  long long n4 = (long long)N * CK / 4;
  init_agg<<<(unsigned)((n4 + 255) / 256), 256, 0, stream>>>((uint4*)agg, n4);

  edge_mlp<<<(unsigned)((totRows + TILE_M - 1) / TILE_M), BLOCK, 0, stream>>>(
      x, ei, E, totRows, Wp, Wp + 65536,
      b1, g1, be1, m1, v1, b2, g2, be2, m2, v2, agg);

  final_mlp<<<(N + TILE_M - 1) / TILE_M, BLOCK, 0, stream>>>(
      agg, Wp + 2 * 65536, b3, g3, be3, m3, v3, (float*)d_out, N);
}
